// GATv2_68728066670717
// MI455X (gfx1250) — matrix operand order repaired and checked by reading, not compiled, not run on hardware
//
#include <hip/hip_runtime.h>

#define N_NODES 50000
#define N_EDGES 1000000
#define E_TOT   (N_EDGES + N_NODES)
#define D_INPUT 128
#define HID     64
#define N_GRAPHS 256

typedef __attribute__((ext_vector_type(16))) __bf16 v16bf;
typedef __attribute__((ext_vector_type(8)))  float  v8f;
typedef unsigned int u32x4 __attribute__((ext_vector_type(4)));
typedef int          i32x8 __attribute__((ext_vector_type(8)));
typedef int          i32x4 __attribute__((ext_vector_type(4)));

// ---------- helpers ----------
__device__ __forceinline__ __bf16 f2bf(float f) { return (__bf16)f; }  // RTNE fptrunc -> hw cvt

// order-preserving float <-> uint encoding (for atomicMax segment-max)
__device__ __forceinline__ unsigned fenc(float f) {
  unsigned u = __float_as_uint(f);
  return (u & 0x80000000u) ? ~u : (u | 0x80000000u);
}
__device__ __forceinline__ float fdec(unsigned u) {
  unsigned v = (u & 0x80000000u) ? (u & 0x7fffffffu) : ~u;
  return __uint_as_float(v);
}

// CDNA5 async memory->LDS copy (no VGPR data path, tracked by ASYNCcnt).
__device__ __forceinline__ void async_b128(unsigned lds_off, const void* gaddr) {
  asm volatile("global_load_async_to_lds_b128 %0, %1, off"
               :: "v"(lds_off), "v"(gaddr) : "memory");
}
__device__ __forceinline__ void wait_async_le4() {
  asm volatile("s_wait_asynccnt 4" ::: "memory");
}
__device__ __forceinline__ void wait_async_le0() {
  asm volatile("s_wait_asynccnt 0" ::: "memory");
}

// CDNA5 Tensor Data Mover: 1-D descriptor copy of `bytes` (multiple of 8) from
// global `W` into LDS offset `ldsW`. count=1, data_size=8B, type=2 ("image"),
// workgroup_mask=0 (not in a cluster). Tracked by TENSORcnt.
// This toolchain exposes the 6-arg builtin: (u32x4, i32x8, i32x4, i32x4, i32x8, i32 cpol).
__device__ __forceinline__ void tdm_load_weights(const void* W, unsigned ldsW, int bytes) {
  unsigned long long ga = (unsigned long long)(size_t)W;
  unsigned d0 = (unsigned)(bytes >> 3);                 // elements of 8 bytes (<= 2048 here)
  u32x4 g0;
  g0[0] = 1u;                                           // count=1, is_restore=0, gather off
  g0[1] = ldsW;                                         // lds_addr (bytes)
  g0[2] = (unsigned)(ga & 0xffffffffu);                 // global_addr[31:0]
  g0[3] = (unsigned)((ga >> 32) & 0x01ffffffu) | (2u << 30); // global_addr[56:32] | type=2
  i32x8 g1;
  g1[0] = (int)(3u << 16);                              // data_size = 3 (8B); mask/flags = 0
  g1[1] = (int)((d0 & 0xffffu) << 16);                  // tensor_dim0[15:0]
  g1[2] = (int)(((d0 >> 16) & 0xffffu) | (1u << 16));   // tensor_dim0[31:16] | tensor_dim1 = 1
  g1[3] = (int)(d0 << 16);                              // tile_dim0 = d0
  g1[4] = 0;                                            // tile_dim1 = tile_dim2 = 0 (unused)
  g1[5] = (int)d0;                                      // tensor_dim0_stride
  g1[6] = 0;
  g1[7] = 0;
  i32x4 z4 = {0, 0, 0, 0};
  i32x8 z8 = {0, 0, 0, 0, 0, 0, 0, 0};
  __builtin_amdgcn_tensor_load_to_lds(g0, g1, z4, z4, z8, 0);
}

// ---------- weight f32 -> bf16 ----------
__global__ void cvt_bf16_kernel(const float* __restrict__ s, __bf16* __restrict__ d, int n) {
  int i = blockIdx.x * blockDim.x + threadIdx.x;
  if (i < n) d[i] = f2bf(s[i]);
}

// ---------- C[N,64] = A[N,K] * W[64,K]^T (+bias), bf16 WMMA, f32 accumulate ----------
// block = 128 threads = 4 waves; block tile = 64 rows x 64 cols; wave tile = 16 rows x 64 cols.
// Weights arrive via one TDM tensor_load_to_lds (TENSORcnt); A chunks (64 rows x 32 K, f32)
// are double-buffered via global_load_async_to_lds_b128 (ASYNCcnt), with chunk c+2 issued
// into the buffer just consumed while chunk c+1's copy is still in flight.
__global__ __launch_bounds__(128) void gemm_wmma(const float* __restrict__ A,
                                                 const __bf16* __restrict__ W,
                                                 const float* __restrict__ bias,
                                                 float* __restrict__ C, int N, int K) {
  __shared__ __bf16 sW[HID * D_INPUT];   // up to 64x128 bf16 = 16 KB
  __shared__ float  sA[2][64 * 32];      // two 8 KB f32 staging buffers
  const int tid  = threadIdx.x;
  const int wave = tid >> 5, lane = tid & 31;
  const int m = lane & 15, half = lane >> 4;
  const int blockRow = blockIdx.x * 64;
  const unsigned ldsW  = (unsigned)(size_t)(void*)sW;
  const unsigned ldsA0 = (unsigned)(size_t)(void*)&sA[0][0];
  const unsigned ldsA1 = (unsigned)(size_t)(void*)&sA[1][0];

  // one wave kicks the TDM weight copy (EXEC-independent DMA, done once per block)
  if (wave == 0) tdm_load_weights(W, ldsW, HID * K * 2);

  // issue one A chunk: 64 rows x 32 f32 = 512 b128 transfers = 4 async instrs per wave
  auto issue_chunk = [&](int k0, unsigned ldsBase) {
#pragma unroll
    for (int u = 0; u < 4; ++u) {
      int i = tid + u * 128;
      int r = i >> 3, seg = i & 7;                      // 8 x b128 per row
      int gr = blockRow + r;
      if (gr > N - 1) gr = N - 1;                       // clamp: branchless OOB handling
      async_b128(ldsBase + (unsigned)(r * 128 + seg * 16),
                 A + (size_t)gr * K + k0 + seg * 4);
    }
  };
  const int nchunks = K >> 5;
  issue_chunk(0, ldsA0);
  if (nchunks > 1) issue_chunk(32, ldsA1);

  // weight DMA must land before the first barrier publishes sW
  if (wave == 0) __builtin_amdgcn_s_wait_tensorcnt(0);

  v8f acc[4] = {};
  for (int c = 0; c < nchunks; ++c) {
    // chunk c complete when at most one later chunk (4 async ops) is outstanding
    if (c + 1 < nchunks) wait_async_le4(); else wait_async_le0();
    __syncthreads();                                    // all copies landed & visible
    const float* sa = sA[c & 1];
    const int k0 = c * 32;
    // A fragment: 16x32 bf16.  VGPR j<4: K = half*8 + 2j (+1); j>=4: K = 16 + half*8 + 2(j-4)
    v16bf a;
#pragma unroll
    for (int j = 0; j < 8; ++j) {
      int bk = (j < 4) ? (half * 8 + 2 * j) : (16 + half * 8 + 2 * (j - 4));
      float2 p = *(const float2*)(sa + m * 32 + bk);    // ds_load_b64
      a[2 * j]     = f2bf(p.x);
      a[2 * j + 1] = f2bf(p.y);
    }
#pragma unroll
    for (int t = 0; t < 4; ++t) {
      // B fragment: 32x16 bf16, B[k][n] = W[n][k]. same K order as A: j<4: K = half*8 + 2j (+1); j>=4: K = 16 + half*8 + 2(j-4)
      int n = t * 16 + m;
      v16bf b;
#pragma unroll
      for (int j = 0; j < 8; ++j) {
        int bk = (j < 4) ? (half * 8 + 2 * j) : (16 + half * 8 + 2 * (j - 4));
        b[2 * j]     = sW[n * K + k0 + bk];
        b[2 * j + 1] = sW[n * K + k0 + bk + 1];
      }
      acc[t] = __builtin_amdgcn_wmma_f32_16x16x32_bf16(false, a, false, b,
                                                       (short)0, acc[t], false, false);
    }
    __syncthreads();                                    // all waves done reading buf[c&1]
    if (c + 2 < nchunks) issue_chunk((c + 2) * 32, (c & 1) ? ldsA1 : ldsA0);
  }
  // C/D layout: VGPR r -> M = r (lanes 0-15) or 8+r (lanes 16-31); N = lane & 15
  const int rowBase = blockRow + wave * 16;
#pragma unroll
  for (int t = 0; t < 4; ++t) {
#pragma unroll
    for (int r = 0; r < 8; ++r) {
      int row = rowBase + ((lane < 16) ? r : (8 + r));
      int col = t * 16 + m;
      if (row < N) {
        float v = acc[t][r];
        if (bias) v += bias[col];
        C[(size_t)row * HID + col] = v;
      }
    }
  }
}

// ---------- per-layer init ----------
__global__ void init_layer_kernel(unsigned* __restrict__ m_u, float* __restrict__ den,
                                  float* __restrict__ agg) {
  int i = blockIdx.x * blockDim.x + threadIdx.x;
  if (i < N_NODES * HID) agg[i] = 0.f;
  if (i < N_NODES) { m_u[i] = 0u; den[i] = 0.f; }
}

// ---------- edge phase 1: logits + segment max (wave per edge) ----------
__global__ void edge_logits_kernel(const float* __restrict__ xl, const float* __restrict__ xr,
                                   const int* __restrict__ srcA, const int* __restrict__ dstA,
                                   const float* __restrict__ att, float* __restrict__ logits,
                                   unsigned* __restrict__ m_u) {
  int wave = threadIdx.x >> 5, lane = threadIdx.x & 31;
  int e = blockIdx.x * (blockDim.x >> 5) + wave;
  if (e >= E_TOT) return;
  int src, dst;
  if (e < N_EDGES) { src = srcA[e]; dst = dstA[e]; } else { src = dst = e - N_EDGES; }
  float s0 = xl[(size_t)src * HID + lane]      + xr[(size_t)dst * HID + lane];
  float s1 = xl[(size_t)src * HID + lane + 32] + xr[(size_t)dst * HID + lane + 32];
  s0 = (s0 > 0.f) ? s0 : 0.2f * s0;
  s1 = (s1 > 0.f) ? s1 : 0.2f * s1;
  float p = s0 * att[lane] + s1 * att[lane + 32];
#pragma unroll
  for (int o = 16; o > 0; o >>= 1) p += __shfl_xor(p, o, 32);
  if (lane == 0) {
    logits[e] = p;
    atomicMax(&m_u[dst], fenc(p));
  }
}

// ---------- edge phase 2: exp + segment sum (thread per edge) ----------
__global__ void edge_exp_kernel(const float* __restrict__ logits, const int* __restrict__ dstA,
                                const unsigned* __restrict__ m_u, float* __restrict__ exb,
                                float* __restrict__ den) {
  int e = blockIdx.x * blockDim.x + threadIdx.x;
  if (e >= E_TOT) return;
  int dst = (e < N_EDGES) ? dstA[e] : (e - N_EDGES);
  float ex = __expf(logits[e] - fdec(m_u[dst]));
  exb[e] = ex;
  atomicAdd(&den[dst], ex);
}

// ---------- edge phase 3: alpha-weighted scatter (wave per edge) ----------
__global__ void edge_scatter_kernel(const float* __restrict__ xl, const int* __restrict__ srcA,
                                    const int* __restrict__ dstA, const float* __restrict__ exb,
                                    const float* __restrict__ den, float* __restrict__ agg) {
  int wave = threadIdx.x >> 5, lane = threadIdx.x & 31;
  int e = blockIdx.x * (blockDim.x >> 5) + wave;
  if (e >= E_TOT) return;
  int src, dst;
  if (e < N_EDGES) { src = srcA[e]; dst = dstA[e]; } else { src = dst = e - N_EDGES; }
  float alpha = exb[e] / den[dst];
  atomicAdd(&agg[(size_t)dst * HID + lane],      alpha * xl[(size_t)src * HID + lane]);
  atomicAdd(&agg[(size_t)dst * HID + lane + 32], alpha * xl[(size_t)src * HID + lane + 32]);
}

// ---------- node epilogue: agg + b + residual, optional ReLU ----------
__global__ void node_epilogue_kernel(const float* __restrict__ agg, const float* __restrict__ res,
                                     const float* __restrict__ b, float* __restrict__ out,
                                     int doRelu) {
  int i = blockIdx.x * blockDim.x + threadIdx.x;
  if (i >= N_NODES * HID) return;
  float v = agg[i] + res[i] + b[i & (HID - 1)];
  out[i] = (doRelu && v < 0.f) ? 0.f : v;
}

// ---------- pooling ----------
__global__ void pool_init_kernel(float* __restrict__ pooled, float* __restrict__ cnt) {
  int i = blockIdx.x * blockDim.x + threadIdx.x;
  if (i < N_GRAPHS * HID) pooled[i] = 0.f;
  if (i < N_GRAPHS) cnt[i] = 0.f;
}
__global__ void pool_accum_kernel(const float* __restrict__ x, const int* __restrict__ batch,
                                  float* __restrict__ pooled, float* __restrict__ cnt) {
  int i = blockIdx.x * blockDim.x + threadIdx.x;
  if (i >= N_NODES * HID) return;
  int n = i >> 6, c = i & (HID - 1);
  int g = batch[n];
  atomicAdd(&pooled[(size_t)g * HID + c], x[i]);
  if (c == 0) atomicAdd(&cnt[g], 1.f);
}
__global__ void final_kernel(const float* __restrict__ pooled, const float* __restrict__ cnt,
                             const float* __restrict__ Wf, const float* __restrict__ bf_,
                             float* __restrict__ out) {
  int g = blockIdx.x, lane = threadIdx.x;
  float c = cnt[g]; c = (c > 1.f) ? c : 1.f;
  float s = pooled[(size_t)g * HID + lane] * Wf[lane] +
            pooled[(size_t)g * HID + lane + 32] * Wf[lane + 32];
#pragma unroll
  for (int o = 16; o > 0; o >>= 1) s += __shfl_xor(s, o, 32);
  if (lane == 0) out[g] = s / c + bf_[0];
}

// ---------- host ----------
extern "C" void kernel_launch(void* const* d_in, const int* in_sizes, int n_in,
                              void* d_out, int out_size, void* d_ws, size_t ws_size,
                              hipStream_t stream) {
  const float* x0    = (const float*)d_in[0];
  const int*   ei    = (const int*)d_in[1];
  const int*   batch = (const int*)d_in[2];
  const float* Wl[3] = {(const float*)d_in[3], (const float*)d_in[9],  (const float*)d_in[15]};
  const float* Wr[3] = {(const float*)d_in[4], (const float*)d_in[10], (const float*)d_in[16]};
  const float* at[3] = {(const float*)d_in[5], (const float*)d_in[11], (const float*)d_in[17]};
  const float* bb[3] = {(const float*)d_in[6], (const float*)d_in[12], (const float*)d_in[18]};
  const float* Rw[3] = {(const float*)d_in[7], (const float*)d_in[13], (const float*)d_in[19]};
  const float* Rb[3] = {(const float*)d_in[8], (const float*)d_in[14], (const float*)d_in[20]};
  const float* Wf  = (const float*)d_in[21];
  const float* bf_ = (const float*)d_in[22];

  char* p = (char*)d_ws;
  auto carve = [&](size_t bytes) { char* r = p; p += (bytes + 255) & ~(size_t)255; return r; };
  const size_t XN = (size_t)N_NODES * HID;
  float*    xcur   = (float*)carve(XN * 4);
  float*    xl     = (float*)carve(XN * 4);
  float*    xr     = (float*)carve(XN * 4);
  float*    res    = (float*)carve(XN * 4);
  float*    agg    = (float*)carve(XN * 4);
  float*    logits = (float*)carve((size_t)E_TOT * 4);
  float*    exb    = (float*)carve((size_t)E_TOT * 4);
  unsigned* m_u    = (unsigned*)carve((size_t)N_NODES * 4);
  float*    den    = (float*)carve((size_t)N_NODES * 4);
  float*    pooled = (float*)carve((size_t)N_GRAPHS * HID * 4);
  float*    cnt    = (float*)carve((size_t)N_GRAPHS * 4);
  __bf16*   wb     = (__bf16*)carve((size_t)49152 * 2);

  int kdim[3] = {D_INPUT, HID, HID};
  __bf16 *WlB[3], *WrB[3], *RwB[3];
  size_t off = 0;
  for (int l = 0; l < 3; ++l) {
    size_t sz = (size_t)HID * kdim[l];
    WlB[l] = wb + off; off += sz;
    WrB[l] = wb + off; off += sz;
    RwB[l] = wb + off; off += sz;
  }
  for (int l = 0; l < 3; ++l) {
    int n = HID * kdim[l];
    int blocks = (n + 255) / 256;
    cvt_bf16_kernel<<<blocks, 256, 0, stream>>>(Wl[l], WlB[l], n);
    cvt_bf16_kernel<<<blocks, 256, 0, stream>>>(Wr[l], WrB[l], n);
    cvt_bf16_kernel<<<blocks, 256, 0, stream>>>(Rw[l], RwB[l], n);
  }

  const int gemmBlocks       = (N_NODES + 63) / 64;
  const int edgeWaveBlocks   = (E_TOT + 7) / 8;      // 8 waves (256 threads) per block
  const int edgeThreadBlocks = (E_TOT + 255) / 256;
  const int nodeChBlocks     = ((int)XN + 255) / 256;
  const int* srcA = ei;
  const int* dstA = ei + N_EDGES;

  const float* Ain = x0;
  for (int l = 0; l < 3; ++l) {
    int K = kdim[l];
    gemm_wmma<<<gemmBlocks, 128, 0, stream>>>(Ain, WlB[l], nullptr, xl, N_NODES, K);
    gemm_wmma<<<gemmBlocks, 128, 0, stream>>>(Ain, WrB[l], nullptr, xr, N_NODES, K);
    gemm_wmma<<<gemmBlocks, 128, 0, stream>>>(Ain, RwB[l], Rb[l],   res, N_NODES, K);
    init_layer_kernel<<<nodeChBlocks, 256, 0, stream>>>(m_u, den, agg);
    edge_logits_kernel<<<edgeWaveBlocks, 256, 0, stream>>>(xl, xr, srcA, dstA, at[l], logits, m_u);
    edge_exp_kernel<<<edgeThreadBlocks, 256, 0, stream>>>(logits, dstA, m_u, exb, den);
    edge_scatter_kernel<<<edgeWaveBlocks, 256, 0, stream>>>(xl, srcA, dstA, exb, den, agg);
    node_epilogue_kernel<<<nodeChBlocks, 256, 0, stream>>>(agg, res, bb[l], xcur, (l < 2) ? 1 : 0);
    Ain = xcur;
  }

  pool_init_kernel<<<(N_GRAPHS * HID + 255) / 256, 256, 0, stream>>>(pooled, cnt);
  pool_accum_kernel<<<nodeChBlocks, 256, 0, stream>>>(xcur, batch, pooled, cnt);
  final_kernel<<<N_GRAPHS, 32, 0, stream>>>(pooled, cnt, Wf, bf_, (float*)d_out);
}
